// ConvLSTMEEGAutoencoder_72370198938012
// MI455X (gfx1250) — matrix operand order repaired and checked by reading, not compiled, not run on hardware
//
#include <hip/hip_runtime.h>
#include <hip/hip_bf16.h>
#include <math.h>
#include <stdint.h>

typedef _Float16 v16h __attribute__((ext_vector_type(16)));
typedef _Float16 v8h  __attribute__((ext_vector_type(8)));
typedef float    v8f  __attribute__((ext_vector_type(8)));

#define Bn 32
#define Cn 64
#define Tn 1024
#define Hn 128
#define HD 32
#define ROWS (Bn*Tn)            // 32768

// ---------------- WMMA tile loaders (CDNA5 16x16x32 f16 layouts) ----------------
// A tile 16x32 (MxK), row-major src: lane<16 -> M=lane, K {0..15};
// lane>=16 -> M=lane-16, K {16..31}.
__device__ inline v16h load_a_tile(const _Float16* __restrict__ src, int ld, int lane) {
    int m  = lane & 15;
    int kb = (lane >> 4) * 16;
    v8h lo = *(const v8h*)(src + (size_t)m * ld + kb);
    v8h hi = *(const v8h*)(src + (size_t)m * ld + kb + 8);
    v16h a;
#pragma unroll
    for (int j = 0; j < 8; ++j) { a[j] = lo[j]; a[8 + j] = hi[j]; }
    return a;
}
// B tile 32x16 (KxN) where B = W^T, W row-major [N x K]:
// lane holds column N=lane&15 (= W row), K = 16*(lane>>4)+j (contiguous 16 halfs).
__device__ inline v16h load_b_tile(const _Float16* __restrict__ W, int ld, int lane) {
    int n  = lane & 15;
    int kb = (lane >> 4) * 16;
    return *(const v16h*)(W + (size_t)n * ld + kb);
}

__device__ inline float sigmoidf_(float x) { return 1.0f / (1.0f + expf(-x)); }

// CDNA5 async copy global->LDS (ASYNCcnt-tracked), GVS addressing: saddr(u64)+vaddr(i32)
__device__ inline void async_copy16(unsigned lds_addr, unsigned voff, uint64_t sbase) {
    asm volatile("global_load_async_to_lds_b128 %0, %1, %2"
                 :: "v"(lds_addr), "v"(voff), "s"(sbase) : "memory");
}
__device__ inline void wait_asynccnt0() {
    asm volatile("s_wait_asynccnt 0" ::: "memory");
}

// ---------------- prep: f32 weights -> f16, bias sums ----------------
__global__ void prep_kernel(const float* eWih, const float* eWhh, const float* ebih,
                            const float* ebhh, const float* dWih, const float* dWhh,
                            const float* dbih, const float* dbhh,
                            _Float16* eWihH, _Float16* eWhhH, _Float16* dWihH,
                            _Float16* dWhhH, float* bsumE, float* bsumD) {
    int i = blockIdx.x * 256 + threadIdx.x;
    if (i < 4 * Hn * Hn)      eWhhH[i] = (_Float16)eWhh[i];     // 512x128
    if (i < 4 * Hn * 32)      eWihH[i] = (_Float16)eWih[i];     // 512x32
    if (i < 4 * HD * Hn)      dWihH[i] = (_Float16)dWih[i];     // 128x128
    if (i < 4 * HD * HD)      dWhhH[i] = (_Float16)dWhh[i];     // 128x32
    if (i < 4 * Hn)           bsumE[i] = ebih[i] + ebhh[i];
    if (i < 4 * HD)           bsumD[i] = dbih[i] + dbhh[i];
}

// ---------------- conv1d(64->32,k3,p1)+ReLU, output [B*T, 32] f16 ----------------
__global__ __launch_bounds__(256) void conv_relu_kernel(const float* __restrict__ x,
                                                        const float* __restrict__ w,
                                                        const float* __restrict__ bias,
                                                        _Float16* __restrict__ hh) {
    __shared__ float wS[32 * 64 * 3];
    __shared__ float bS[32];
    for (int i = threadIdx.x; i < 32 * 64 * 3; i += 256) wS[i] = w[i];
    if (threadIdx.x < 32) bS[threadIdx.x] = bias[threadIdx.x];
    __syncthreads();
    int gid = blockIdx.x * 256 + threadIdx.x;   // (b,t)
    int b = gid >> 10, t = gid & 1023;
    float acc[32];
#pragma unroll
    for (int co = 0; co < 32; ++co) acc[co] = bS[co];
    for (int ci = 0; ci < 64; ++ci) {
        const float* xr = x + ((size_t)b * 64 + ci) * Tn;
        float x0 = (t > 0)    ? xr[t - 1] : 0.f;
        float x1 = xr[t];
        float x2 = (t < 1023) ? xr[t + 1] : 0.f;
#pragma unroll
        for (int co = 0; co < 32; ++co) {
            const float* wp = wS + (co * 64 + ci) * 3;
            acc[co] += wp[0] * x0 + wp[1] * x1 + wp[2] * x2;
        }
    }
#pragma unroll
    for (int co = 0; co < 32; ++co)
        hh[(size_t)gid * 32 + co] = (_Float16)fmaxf(acc[co], 0.f);
}

// ---------------- generic WMMA GEMM: out[M,N] = A[M,K] * W[N,K]^T + bias ----------------
template <typename OT>
__global__ __launch_bounds__(256, 1) void wmma_gemm_bias(const _Float16* __restrict__ A, int lda,
                                                         const _Float16* __restrict__ W, int ldw,
                                                         const float* __restrict__ bias,
                                                         OT* __restrict__ out, int ldo,
                                                         int ntn, int kchunks) {
    int lane = threadIdx.x & 31, wave = threadIdx.x >> 5;
    long tile = (long)blockIdx.x * 8 + wave;
    int mt = (int)(tile / ntn), nt = (int)(tile % ntn);
    const _Float16* Ap = A + (size_t)mt * 16 * lda;
    const _Float16* Wp = W + (size_t)nt * 16 * ldw;
    v8f acc = {};
    for (int kc = 0; kc < kchunks; ++kc) {
        v16h a = load_a_tile(Ap + kc * 32, lda, lane);
        v16h b = load_b_tile(Wp + kc * 32, ldw, lane);
        acc = __builtin_amdgcn_wmma_f32_16x16x32_f16(false, a, false, b, (short)0, acc, false, false);
    }
    int n = nt * 16 + (lane & 15);
    float bv = bias ? bias[n] : 0.f;
#pragma unroll
    for (int r = 0; r < 8; ++r) {
        int m = mt * 16 + 8 * (lane >> 4) + r;
        out[(size_t)m * ldo + n] = (OT)(acc[r] + bv);
    }
}

// ---------------- encoder LSTM scan ----------------
// grid=2 (16 batch rows each), block=256 (8 waves). Whh register-resident;
// gx[t] double-buffered in LDS via async global->LDS copies (ASYNCcnt).
__global__ __launch_bounds__(256, 1) void enc_scan_kernel(const _Float16* __restrict__ gx,
                                                          const _Float16* __restrict__ Whh,
                                                          float* __restrict__ hout) {
    __shared__ __align__(32) _Float16 hS[16 * 136];      // padded ld (bank-conflict free)
    __shared__ float cS[16 * 128];
    __shared__ __align__(16) _Float16 gxS[2][16 * 512];  // double-buffered gate slab
    int lane = threadIdx.x & 31, wave = threadIdx.x >> 5;
    int b0 = blockIdx.x * 16;
    uint64_t gbase = (uint64_t)(uintptr_t)gx;
    int arow = threadIdx.x >> 4;        // 0..15: row this thread copies
    int aseg = threadIdx.x & 15;        // 4 x 16B chunks each (row = 1024B)

    // Whh f16 [512 x 128] -> 16 register-resident B tiles per wave (all 1024 steps)
    v16h Bt[4][4];
#pragma unroll
    for (int nt = 0; nt < 4; ++nt) {
        int n0 = (wave * 4 + nt) * 16;
#pragma unroll
        for (int kc = 0; kc < 4; ++kc)
            Bt[nt][kc] = load_b_tile(Whh + (size_t)n0 * 128 + kc * 32, 128, lane);
    }
    for (int i = threadIdx.x; i < 16 * 136; i += 256) hS[i] = (_Float16)0;
    for (int i = threadIdx.x; i < 16 * 128; i += 256) cS[i] = 0.f;
    // prologue: bring gx[0] into buffer 0
#pragma unroll
    for (int j = 0; j < 4; ++j) {
        int chunk = aseg * 4 + j;
        unsigned voff = (unsigned)(((((b0 + arow) * Tn) + 0) * 512 + chunk * 8) * 2);
        unsigned lds  = (unsigned)(uintptr_t)&gxS[0][arow * 512 + chunk * 8];
        async_copy16(lds, voff, gbase);
    }
    wait_asynccnt0();
    __syncthreads();

    for (int t = 0; t < Tn; ++t) {
        if (t + 1 < Tn) {   // overlap next step's gx fetch with this step's WMMAs
            int nb = (t + 1) & 1;
#pragma unroll
            for (int j = 0; j < 4; ++j) {
                int chunk = aseg * 4 + j;
                unsigned voff = (unsigned)(((((b0 + arow) * Tn) + (t + 1)) * 512 + chunk * 8) * 2);
                unsigned lds  = (unsigned)(uintptr_t)&gxS[nb][arow * 512 + chunk * 8];
                async_copy16(lds, voff, gbase);
            }
        }
        _Float16* gcur = gxS[t & 1];
        v16h At[4];
#pragma unroll
        for (int kc = 0; kc < 4; ++kc) At[kc] = load_a_tile(hS + kc * 32, 136, lane);
#pragma unroll
        for (int nt = 0; nt < 4; ++nt) {
            v8f acc = {};
#pragma unroll
            for (int kc = 0; kc < 4; ++kc)
                acc = __builtin_amdgcn_wmma_f32_16x16x32_f16(false, At[kc], false, Bt[nt][kc],
                                                             (short)0, acc, false, false);
            int col = (wave * 4 + nt) * 16 + (lane & 15);
#pragma unroll
            for (int r = 0; r < 8; ++r) {           // in-place: g = h*Whh^T + gx
                int m = 8 * (lane >> 4) + r;
                gcur[m * 512 + col] = (_Float16)(acc[r] + (float)gcur[m * 512 + col]);
            }
        }
        __syncthreads();
#pragma unroll
        for (int e = 0; e < 8; ++e) {
            int idx = threadIdx.x * 8 + e;          // 0..2047
            int m = idx >> 7, hc = idx & 127;
            const _Float16* gr = gcur + m * 512;
            float ig = sigmoidf_((float)gr[hc]);
            float fg = sigmoidf_((float)gr[128 + hc]);
            float gg = tanhf((float)gr[256 + hc]);
            float og = sigmoidf_((float)gr[384 + hc]);
            float c  = fg * cS[m * 128 + hc] + ig * gg;
            cS[m * 128 + hc] = c;
            float h = og * tanhf(c);
            hS[m * 136 + hc] = (_Float16)h;
            hout[((size_t)(b0 + m) * Tn + t) * 128 + hc] = h;
        }
        wait_asynccnt0();      // next buffer resident before anyone reads it
        __syncthreads();
    }
}

// ---------------- row normalize + f16 copies ----------------
__global__ __launch_bounds__(256) void normalize_kernel(const float* __restrict__ lo,
                                                        _Float16* __restrict__ lh,
                                                        _Float16* __restrict__ nvh) {
    int lane = threadIdx.x & 31, wv = threadIdx.x >> 5;
    int row = blockIdx.x * 8 + wv;
    const float* r = lo + (size_t)row * 128;
    float v[4]; float ss = 0.f;
#pragma unroll
    for (int j = 0; j < 4; ++j) { v[j] = r[lane * 4 + j]; ss += v[j] * v[j]; }
#pragma unroll
    for (int off = 16; off; off >>= 1) ss += __shfl_xor(ss, off, 32);
    float inv = rsqrtf(ss);
#pragma unroll
    for (int j = 0; j < 4; ++j) {
        lh [(size_t)row * 128 + lane * 4 + j] = (_Float16)v[j];
        nvh[(size_t)row * 128 + lane * 4 + j] = (_Float16)(v[j] * inv);
    }
}

// ---------------- recurrence matrix: WMMA cos = nv*nv^T, clamp/compare ----------------
// acos(clip(c)) <= eps  <=>  clip(c) >= cos(eps)   (acos monotone decreasing)
__global__ __launch_bounds__(256, 1) void rec_kernel(const _Float16* __restrict__ nvh,
                                                     const float* __restrict__ epsp,
                                                     float* __restrict__ rec_out) {
    int lane = threadIdx.x & 31, wave = threadIdx.x >> 5;
    int b = blockIdx.x >> 9;                    // 512 blocks per batch
    int tile = (blockIdx.x & 511) * 8 + wave;   // 0..4095 -> 64x64 tiles
    int ts = tile >> 6, ss = tile & 63;
    const _Float16* base = nvh + (size_t)b * Tn * 128;
    const _Float16* Ap = base + (size_t)ts * 16 * 128;
    const _Float16* Bp = base + (size_t)ss * 16 * 128;  // B = nv^T: column n = nv row n
    v8f acc = {};
#pragma unroll
    for (int kc = 0; kc < 4; ++kc) {
        v16h a  = load_a_tile(Ap + kc * 32, 128, lane);
        v16h bm = load_b_tile(Bp + kc * 32, 128, lane);
        acc = __builtin_amdgcn_wmma_f32_16x16x32_f16(false, a, false, bm, (short)0, acc, false, false);
    }
    float ce = cosf(fminf(fmaxf(*epsp, 0.f), 3.14159265f));
    int n = ss * 16 + (lane & 15);
    float* out = rec_out + (size_t)b * Tn * Tn;
#pragma unroll
    for (int r = 0; r < 8; ++r) {
        int m = ts * 16 + 8 * (lane >> 4) + r;
        float c = fminf(fmaxf(acc[r], -1.f + 1e-7f), 1.f - 1e-7f);
        out[(size_t)m * Tn + n] = (c >= ce) ? 1.f : 0.f;
    }
}

// ---------------- decoder LSTM scan (H=32): async gx + 1 WMMA/wave/step ----------------
__global__ __launch_bounds__(256, 1) void dec_scan_kernel(const _Float16* __restrict__ gx,
                                                          const _Float16* __restrict__ Whh,
                                                          float* __restrict__ dout) {
    __shared__ __align__(32) _Float16 hS[16 * 40];       // padded ld
    __shared__ float cS[16 * 32];
    __shared__ __align__(16) _Float16 gxS[2][16 * 128];  // double-buffered gate slab
    int lane = threadIdx.x & 31, wave = threadIdx.x >> 5;
    int b0 = blockIdx.x * 16;
    int n0 = wave * 16;
    uint64_t gbase = (uint64_t)(uintptr_t)gx;
    int arow = threadIdx.x >> 4;   // 0..15
    int aseg = threadIdx.x & 15;   // one 16B chunk each (row = 256B)
    v16h Bt = load_b_tile(Whh + (size_t)n0 * 32, 32, lane);   // [128 x 32], K=32
    for (int i = threadIdx.x; i < 16 * 40; i += 256) hS[i] = (_Float16)0;
    for (int i = threadIdx.x; i < 16 * 32; i += 256) cS[i] = 0.f;
    {
        unsigned voff = (unsigned)(((((b0 + arow) * Tn) + 0) * 128 + aseg * 8) * 2);
        unsigned lds  = (unsigned)(uintptr_t)&gxS[0][arow * 128 + aseg * 8];
        async_copy16(lds, voff, gbase);
    }
    wait_asynccnt0();
    __syncthreads();
    for (int t = 0; t < Tn; ++t) {
        if (t + 1 < Tn) {
            int nb = (t + 1) & 1;
            unsigned voff = (unsigned)(((((b0 + arow) * Tn) + (t + 1)) * 128 + aseg * 8) * 2);
            unsigned lds  = (unsigned)(uintptr_t)&gxS[nb][arow * 128 + aseg * 8];
            async_copy16(lds, voff, gbase);
        }
        _Float16* gcur = gxS[t & 1];
        v16h a = load_a_tile(hS, 40, lane);
        v8f acc = {};
        acc = __builtin_amdgcn_wmma_f32_16x16x32_f16(false, a, false, Bt, (short)0, acc, false, false);
        int col = n0 + (lane & 15);
#pragma unroll
        for (int r = 0; r < 8; ++r) {
            int m = 8 * (lane >> 4) + r;
            gcur[m * 128 + col] = (_Float16)(acc[r] + (float)gcur[m * 128 + col]);
        }
        __syncthreads();
#pragma unroll
        for (int e = 0; e < 2; ++e) {
            int idx = threadIdx.x * 2 + e;     // 0..511
            int m = idx >> 5, hc = idx & 31;
            const _Float16* gr = gcur + m * 128;
            float ig = sigmoidf_((float)gr[hc]);
            float fg = sigmoidf_((float)gr[32 + hc]);
            float gg = tanhf((float)gr[64 + hc]);
            float og = sigmoidf_((float)gr[96 + hc]);
            float c  = fg * cS[m * 32 + hc] + ig * gg;
            cS[m * 32 + hc] = c;
            float h = og * tanhf(c);
            hS[m * 40 + hc] = (_Float16)h;
            dout[((size_t)(b0 + m) * Tn + t) * 32 + hc] = h;
        }
        wait_asynccnt0();
        __syncthreads();
    }
}

// ---------------- deconv (ConvTranspose1d 32->64,k3,p1) + tanh ----------------
__global__ __launch_bounds__(256) void deconv_tanh_kernel(const float* __restrict__ d,
                                                          const float* __restrict__ w,   // [32,64,3]
                                                          const float* __restrict__ bias,
                                                          float* __restrict__ out) {
    __shared__ float wS[32 * 64 * 3];
    __shared__ float bS[64];
    for (int i = threadIdx.x; i < 32 * 64 * 3; i += 256) wS[i] = w[i];
    if (threadIdx.x < 64) bS[threadIdx.x] = bias[threadIdx.x];
    __syncthreads();
    int gid = blockIdx.x * 256 + threadIdx.x;
    int b = gid >> 10, t = gid & 1023;
    float acc[64];
#pragma unroll
    for (int co = 0; co < 64; ++co) acc[co] = bS[co];
    for (int ci = 0; ci < 32; ++ci) {
#pragma unroll
        for (int kk = 0; kk < 3; ++kk) {
            int tt = t + kk - 1;
            if (tt < 0 || tt > 1023) continue;
            float dv = d[((size_t)b * Tn + tt) * 32 + ci];
            // w_conv[co][ci][kk] = deconv_w[ci][co][2-kk]
            const float* wp = wS + (size_t)ci * 64 * 3 + (2 - kk);
            for (int co = 0; co < 64; ++co) acc[co] += dv * wp[co * 3];
        }
    }
    for (int co = 0; co < 64; ++co)
        out[((size_t)b * 64 + co) * Tn + t] = tanhf(acc[co]);
}

// ---------------- host launch ----------------
extern "C" void kernel_launch(void* const* d_in, const int* in_sizes, int n_in,
                              void* d_out, int out_size, void* d_ws, size_t ws_size,
                              hipStream_t stream) {
    const float* x       = (const float*)d_in[0];
    const float* conv_w  = (const float*)d_in[1];
    const float* conv_b  = (const float*)d_in[2];
    const float* eWih    = (const float*)d_in[3];
    const float* eWhh    = (const float*)d_in[4];
    const float* ebih    = (const float*)d_in[5];
    const float* ebhh    = (const float*)d_in[6];
    const float* dWih    = (const float*)d_in[7];
    const float* dWhh    = (const float*)d_in[8];
    const float* dbih    = (const float*)d_in[9];
    const float* dbhh    = (const float*)d_in[10];
    const float* dcv_w   = (const float*)d_in[11];
    const float* dcv_b   = (const float*)d_in[12];
    const float* eps     = (const float*)d_in[13];

    float* recon = (float*)d_out;                         // [B,C,T]
    float* rec   = (float*)d_out + (size_t)Bn * Cn * Tn;  // [B,T,T]

    // workspace carve-up (256B aligned)
    char* ws = (char*)d_ws;
    size_t off = 0;
    auto carve = [&](size_t bytes) { size_t o = off; off = (off + bytes + 255) & ~(size_t)255; return o; };
    _Float16* hh    = (_Float16*)(ws + carve((size_t)ROWS * 32 * 2));    // conv out f16
    _Float16* gxhE  = (_Float16*)(ws + carve((size_t)ROWS * 512 * 2));   // enc gate slab f16
    float*    lout  = (float*)   (ws + carve((size_t)ROWS * 128 * 4));   // lstm_out f32
    _Float16* lh    = (_Float16*)(ws + carve((size_t)ROWS * 128 * 2));   // lstm_out f16
    _Float16* nvh   = (_Float16*)(ws + carve((size_t)ROWS * 128 * 2));   // normalized f16
    _Float16* gxhD  = (_Float16*)(ws + carve((size_t)ROWS * 128 * 2));   // dec gate slab f16
    float*    ddec  = (float*)   (ws + carve((size_t)ROWS * 32 * 4));    // decoder hidden f32
    _Float16* eWihH = (_Float16*)(ws + carve(512 * 32 * 2));
    _Float16* eWhhH = (_Float16*)(ws + carve(512 * 128 * 2));
    _Float16* dWihH = (_Float16*)(ws + carve(128 * 128 * 2));
    _Float16* dWhhH = (_Float16*)(ws + carve(128 * 32 * 2));
    float*    bsumE = (float*)   (ws + carve(512 * 4));
    float*    bsumD = (float*)   (ws + carve(128 * 4));
    (void)ws_size; (void)in_sizes; (void)n_in; (void)out_size;

    prep_kernel<<<256, 256, 0, stream>>>(eWih, eWhh, ebih, ebhh, dWih, dWhh, dbih, dbhh,
                                         eWihH, eWhhH, dWihH, dWhhH, bsumE, bsumD);
    conv_relu_kernel<<<ROWS / 256, 256, 0, stream>>>(x, conv_w, conv_b, hh);
    // gx_enc = hh[32768,32] * eWih[512,32]^T + bsum  (2048 x 32 tiles, K=32), f16 out
    wmma_gemm_bias<_Float16><<<(2048 * 32) / 8, 256, 0, stream>>>(hh, 32, eWihH, 32, bsumE,
                                                                  gxhE, 512, 32, 1);
    enc_scan_kernel<<<2, 256, 0, stream>>>(gxhE, eWhhH, lout);
    normalize_kernel<<<ROWS / 8, 256, 0, stream>>>(lout, lh, nvh);
    // rec = (clip(nv*nv^T) >= cos(eps))   32 batches x 4096 tiles
    rec_kernel<<<32 * 512, 256, 0, stream>>>(nvh, eps, rec);
    // gx_dec = lh[32768,128] * dWih[128,128]^T + bsum  (2048 x 8 tiles, K=128), f16 out
    wmma_gemm_bias<_Float16><<<(2048 * 8) / 8, 256, 0, stream>>>(lh, 128, dWihH, 128, bsumD,
                                                                 gxhD, 128, 8, 4);
    dec_scan_kernel<<<2, 256, 0, stream>>>(gxhD, dWhhH, ddec);
    deconv_tanh_kernel<<<ROWS / 256, 256, 0, stream>>>(ddec, dcv_w, dcv_b, recon);
}
